// LinkGramAttention_89249420411548
// MI455X (gfx1250) — matrix operand order repaired and checked by reading, not compiled, not run on hardware
//
#include <hip/hip_runtime.h>
#include <hip/hip_bf16.h>

#define EMBED_DIM 1024
#define NUM_HEADS 16
#define HEAD_DIM 64
#define MAX_DIST 128
#define NUM_LINKS 64
#define BATCH 2
#define SEQ 2048
#define SCALING 0.125f  // HEAD_DIM^-0.5

typedef __attribute__((ext_vector_type(16))) __bf16 v16bf;
typedef __attribute__((ext_vector_type(8)))  float  v8f;
typedef __attribute__((ext_vector_type(4)))  int    v4i;

union FragU { v16bf v; unsigned u[8]; };

// ---- CDNA5 async global->LDS path (guarded; falls back to sync copies) ----
#if defined(__AMDGCN__) && __has_builtin(__builtin_amdgcn_global_load_async_to_lds_b128) && \
    __has_builtin(__builtin_amdgcn_s_wait_asynccnt)
#define HAVE_ASYNC_LDS 1
#define AS1 __attribute__((address_space(1)))
#define AS3 __attribute__((address_space(3)))
__device__ inline void async_copy16(const unsigned short* g, unsigned short* l) {
  __builtin_amdgcn_global_load_async_to_lds_b128(
      (AS1 v4i*)const_cast<unsigned short*>(g), (AS3 v4i*)l, 0, 0);
}
__device__ inline void wait_async0() { __builtin_amdgcn_s_wait_asynccnt(0); }
#else
#define HAVE_ASYNC_LDS 0
__device__ inline void wait_async0() {}
#endif

__device__ inline v8f zero8() {
  v8f z = {0.f,0.f,0.f,0.f,0.f,0.f,0.f,0.f};
  return z;
}

__device__ inline v8f wmma_bf16(v16bf a, v16bf b, v8f c) {
  // D = A(16x32 bf16) * B(32x16 bf16) + C(16x16 f32)
  return __builtin_amdgcn_wmma_f32_16x16x32_bf16(false, a, false, b, (short)0, c,
                                                 false, false);
}

__device__ inline unsigned short f2bf(float f) {
  __bf16 h = (__bf16)f;
  unsigned short s;
  __builtin_memcpy(&s, &h, 2);
  return s;
}

// A fragment (16x32 bf16) from row-major [rows][ld] storage.
// Lane L: m = L&15, k-base = (L>>4)*8; VGPR i<4: K = kb+2i; i>=4: K = 16+kb+2(i-4).
__device__ inline v16bf load_frag_a(const unsigned short* base, int row0, int ld,
                                    int kbase, int lane) {
  int m = lane & 15, kb = (lane >> 4) * 8;
  const unsigned short* p = base + (size_t)(row0 + m) * ld + kbase;
  FragU f;
#pragma unroll
  for (int i = 0; i < 4; ++i) f.u[i] = *(const unsigned*)(p + kb + 2 * i);
#pragma unroll
  for (int i = 0; i < 4; ++i) f.u[4 + i] = *(const unsigned*)(p + 16 + kb + 2 * i);
  return f.v;
}

// B fragment (32x16 bf16) from K-contiguous (transposed) storage [n][k], ld = k-stride.
// Lane L: n = L&15, k-base = (L>>4)*8; VGPR i<4: K = kb+2i; i>=4: K = 16+kb+2(i-4).
__device__ inline v16bf load_frag_bt(const unsigned short* baseT, int col0, int ld,
                                     int kbase, int lane) {
  int n = lane & 15, kb = (lane >> 4) * 8;
  const unsigned short* p = baseT + (size_t)(col0 + n) * ld + kbase + kb;
  FragU f;
#pragma unroll
  for (int i = 0; i < 8; ++i) f.u[i] = *(const unsigned*)(p + (i >> 2) * 16 + 2 * (i & 3));
  return f.v;
}

// ---------------------------------------------------------------------------
// f32 -> bf16 conversion
// ---------------------------------------------------------------------------
__global__ void cvt_bf16_kernel(const float* __restrict__ src,
                                unsigned short* __restrict__ dst, int n) {
  int i = blockIdx.x * blockDim.x + threadIdx.x;
  if (i < n) dst[i] = f2bf(src[i]);
}

// ---------------------------------------------------------------------------
// Shared GEMM machinery: 128x128 block tile, BK=32, double-buffered LDS,
// 8 waves (4x2 grid of 32x64 sub-tiles), 2x4 WMMA accumulators per wave.
// ---------------------------------------------------------------------------
__device__ inline void mma_tiles(const unsigned short* ldsA, const unsigned short* ldsBt,
                                 int wm, int wn, int lane, v8f acc[2][4]) {
  v16bf af[2], bfr[4];
#pragma unroll
  for (int mt = 0; mt < 2; ++mt)
    af[mt] = load_frag_a(ldsA, wm * 32 + mt * 16, 32, 0, lane);
#pragma unroll
  for (int nt = 0; nt < 4; ++nt)
    bfr[nt] = load_frag_bt(ldsBt, wn * 64 + nt * 16, 32, 0, lane);
#pragma unroll
  for (int mt = 0; mt < 2; ++mt)
#pragma unroll
    for (int nt = 0; nt < 4; ++nt)
      acc[mt][nt] = wmma_bf16(af[mt], bfr[nt], acc[mt][nt]);
}

__device__ inline void gemm_issue_tiles(const unsigned short* __restrict__ A,
                                        const unsigned short* __restrict__ W,
                                        unsigned short* ldsA, unsigned short* ldsBt,
                                        int m0, int n0, int kt, int K, int N, int tid) {
  {  // A tile 128x32 row-major: async DMA straight into LDS when available
    int row = tid >> 1, half = tid & 1;
    const unsigned short* ga = A + (size_t)(m0 + row) * K + kt + half * 16;
    unsigned short* la = &ldsA[row * 32 + half * 16];
#if HAVE_ASYNC_LDS
    async_copy16(ga, la);
    async_copy16(ga + 8, la + 8);
#else
    ((uint4*)la)[0] = ((const uint4*)ga)[0];
    ((uint4*)la)[1] = ((const uint4*)ga)[1];
#endif
  }
  {  // B tile 32x128, stored transposed [n][k] so fragments read dword pairs
    int kr = tid >> 3, seg = tid & 7;
    union { uint4 q[2]; unsigned short s16[16]; } t;
    const uint4* sp = (const uint4*)(W + (size_t)(kt + kr) * N + n0 + seg * 16);
    t.q[0] = sp[0];
    t.q[1] = sp[1];
#pragma unroll
    for (int j = 0; j < 16; ++j) ldsBt[(seg * 16 + j) * 32 + kr] = t.s16[j];
  }
}

// ---------------------------------------------------------------------------
// QKV projection: X[4096,1024] @ {Wq,Wk,Wv} + bias, store bf16 split per head
// [B,H,S,64]; Q pre-scaled by SCALING.
// ---------------------------------------------------------------------------
__global__ __launch_bounds__(256) void qkv_gemm_kernel(
    const unsigned short* __restrict__ X,
    const unsigned short* __restrict__ Wq16, const unsigned short* __restrict__ Wk16,
    const unsigned short* __restrict__ Wv16,
    const float* __restrict__ bq, const float* __restrict__ bk,
    const float* __restrict__ bv,
    unsigned short* __restrict__ qout, unsigned short* __restrict__ kout,
    unsigned short* __restrict__ vout) {
  const int K = EMBED_DIM, N = EMBED_DIM;
  __shared__ __align__(16) unsigned short ldsA[2][128 * 32];
  __shared__ __align__(16) unsigned short ldsBt[2][128 * 32];
  int tid = threadIdx.x, lane = tid & 31, wave = tid >> 5;
  int wm = wave & 3, wn = wave >> 2;
  int m0 = blockIdx.y * 128, n0 = blockIdx.x * 128;
  int z = blockIdx.z;
  const unsigned short* W = (z == 0) ? Wq16 : (z == 1) ? Wk16 : Wv16;
  const float* bias = (z == 0) ? bq : (z == 1) ? bk : bv;
  unsigned short* dst = (z == 0) ? qout : (z == 1) ? kout : vout;
  float alpha = (z == 0) ? SCALING : 1.0f;

  v8f acc[2][4];
#pragma unroll
  for (int i = 0; i < 2; ++i)
#pragma unroll
    for (int j = 0; j < 4; ++j) acc[i][j] = zero8();

  gemm_issue_tiles(X, W, ldsA[0], ldsBt[0], m0, n0, 0, K, N, tid);
  int buf = 0;
  for (int kt = 0; kt < K; kt += 32) {
    wait_async0();       // this wave's async writes for tile `buf` complete
    __syncthreads();     // all waves' writes visible; prior reads of buf^1 done
    if (kt + 32 < K)
      gemm_issue_tiles(X, W, ldsA[buf ^ 1], ldsBt[buf ^ 1], m0, n0, kt + 32, K, N, tid);
    mma_tiles(ldsA[buf], ldsBt[buf], wm, wn, lane, acc);
    buf ^= 1;
  }

#pragma unroll
  for (int mt = 0; mt < 2; ++mt)
#pragma unroll
    for (int nt = 0; nt < 4; ++nt)
#pragma unroll
      for (int r = 0; r < 8; ++r) {
        int m = m0 + wm * 32 + mt * 16 + r + ((lane >> 4) << 3);
        int n = n0 + wn * 64 + nt * 16 + (lane & 15);
        float val = (acc[mt][nt][r] + bias[n]) * alpha;
        int b = m / SEQ, s = m % SEQ;
        int h = n >> 6, hd = n & 63;
        dst[(((size_t)(b * NUM_HEADS + h)) * SEQ + s) * HEAD_DIM + hd] = f2bf(val);
      }
}

// ---------------------------------------------------------------------------
// Flash attention with link/distance bias. One block = (b, h, 128 q rows),
// 8 waves * 16 rows. K/V stream through double-buffered LDS (K via async DMA,
// V transposed-scattered); per-head bias tables in LDS; index matrices
// gathered from global (L2-resident across the 16 heads: 67MB < 192MB L2).
// ---------------------------------------------------------------------------
__device__ inline void flash_issue_kv(const unsigned short* kp, const unsigned short* vp,
                                      unsigned short* ldsK, unsigned short* ldsVt,
                                      int kt, int tid) {
  int kr = tid >> 3, seg = tid & 7;
  const unsigned short* gk = kp + (size_t)(kt + kr) * HEAD_DIM + seg * 8;
#if HAVE_ASYNC_LDS
  async_copy16(gk, &ldsK[kr * 64 + seg * 8]);
#else
  *(uint4*)&ldsK[kr * 64 + seg * 8] = *(const uint4*)gk;
#endif
  union { uint4 q4; unsigned short s16[8]; } t;
  t.q4 = *(const uint4*)(vp + (size_t)(kt + kr) * HEAD_DIM + seg * 8);
#pragma unroll
  for (int j = 0; j < 8; ++j) ldsVt[(seg * 8 + j) * 32 + kr] = t.s16[j];
}

__global__ __launch_bounds__(256) void flash_attn_kernel(
    const unsigned short* __restrict__ qm, const unsigned short* __restrict__ km,
    const unsigned short* __restrict__ vm,
    const float* __restrict__ dtab, const float* __restrict__ ltab,
    const int* __restrict__ distm, const int* __restrict__ linkm,
    unsigned short* __restrict__ attn) {
  __shared__ float l_dist[MAX_DIST + 1];
  __shared__ float l_link[NUM_LINKS];
  __shared__ __align__(16) unsigned short ldsK[2][32 * 64];   // [kpos][d]
  __shared__ __align__(16) unsigned short ldsVt[2][64 * 32];  // [d][kpos]
  __shared__ __align__(16) unsigned short ldsP[8][16 * 32];

  int tid = threadIdx.x, lane = tid & 31, wave = tid >> 5;
  int qt = blockIdx.x, h = blockIdx.y, b = blockIdx.z;

  for (int i = tid; i <= MAX_DIST; i += 256) l_dist[i] = dtab[i * NUM_HEADS + h];
  for (int i = tid; i < NUM_LINKS; i += 256) l_link[i] = ltab[i * NUM_HEADS + h];

  size_t headbase = ((size_t)(b * NUM_HEADS + h)) * SEQ * HEAD_DIM;
  const unsigned short* qp = qm + headbase;
  const unsigned short* kp = km + headbase;
  const unsigned short* vp = vm + headbase;

  int sq0 = qt * 128 + wave * 16;
  v16bf qf[2];
#pragma unroll
  for (int kh = 0; kh < 2; ++kh)
    qf[kh] = load_frag_a(qp + (size_t)sq0 * HEAD_DIM, 0, HEAD_DIM, kh * 32, lane);

  float mrow[8], lrow[8];
  v8f oacc[4];
#pragma unroll
  for (int r = 0; r < 8; ++r) { mrow[r] = -1e30f; lrow[r] = 0.f; }
#pragma unroll
  for (int d = 0; d < 4; ++d) oacc[d] = zero8();

  flash_issue_kv(kp, vp, ldsK[0], ldsVt[0], 0, tid);
  int buf = 0;
  for (int kt = 0; kt < SEQ; kt += 32) {
    wait_async0();       // own async K writes for `buf` complete
    __syncthreads();     // everyone's K/V (and tables, 1st iter) visible
    if (kt + 32 < SEQ)
      flash_issue_kv(kp, vp, ldsK[buf ^ 1], ldsVt[buf ^ 1], kt + 32, tid);

    const unsigned short* curK = ldsK[buf];
    const unsigned short* curV = ldsVt[buf];

    // S = Q * K^T (scaling folded into Q); two 16-col score tiles
    v8f sc[2];
#pragma unroll
    for (int nt = 0; nt < 2; ++nt) {
      sc[nt] = zero8();
#pragma unroll
      for (int kh = 0; kh < 2; ++kh) {
        v16bf bfrag = load_frag_bt(curK, nt * 16, 64, kh * 32, lane);
        sc[nt] = wmma_bf16(qf[kh], bfrag, sc[nt]);
      }
    }

    // gather link/distance bias per score element (C-fragment layout)
#pragma unroll
    for (int nt = 0; nt < 2; ++nt)
#pragma unroll
      for (int r = 0; r < 8; ++r) {
        int m = r + ((lane >> 4) << 3);
        int sq = sq0 + m;
        int sk = kt + nt * 16 + (lane & 15);
        size_t off = ((size_t)b * SEQ + sq) * SEQ + sk;
        int dv = distm[off], lv = linkm[off];
        if (kt + 32 < SEQ && nt == 0 && r == 0) {  // prefetch next tile's rows
          __builtin_prefetch(distm + off + 32, 0, 0);
          __builtin_prefetch(linkm + off + 32, 0, 0);
        }
        float bias = 0.f;
        if (dv >= 0) bias += l_dist[dv > MAX_DIST ? MAX_DIST : dv];
        if (lv != -1)
          bias += l_link[lv < 0 ? 0 : (lv > NUM_LINKS - 1 ? NUM_LINKS - 1 : lv)];
        sc[nt][r] += bias;
      }

    // online softmax; each row lives across 16 lanes of a half-wave
#pragma unroll
    for (int r = 0; r < 8; ++r) {
      float t = fmaxf(sc[0][r], sc[1][r]);
#pragma unroll
      for (int msk = 1; msk <= 8; msk <<= 1) t = fmaxf(t, __shfl_xor(t, msk, 32));
      float mnew = fmaxf(mrow[r], t);
      float al = __expf(mrow[r] - mnew);
      mrow[r] = mnew;
      float p0 = __expf(sc[0][r] - mnew);
      float p1 = __expf(sc[1][r] - mnew);
      sc[0][r] = p0;
      sc[1][r] = p1;
      float rs = p0 + p1;
#pragma unroll
      for (int msk = 1; msk <= 8; msk <<= 1) rs += __shfl_xor(rs, msk, 32);
      lrow[r] = lrow[r] * al + rs;
#pragma unroll
      for (int d = 0; d < 4; ++d) oacc[d][r] *= al;
    }

    // P: C-layout -> A-layout via wave-private LDS round trip
    unsigned short* pw = ldsP[wave];
#pragma unroll
    for (int nt = 0; nt < 2; ++nt)
#pragma unroll
      for (int r = 0; r < 8; ++r) {
        int m = r + ((lane >> 4) << 3);
        pw[m * 32 + nt * 16 + (lane & 15)] = f2bf(sc[nt][r]);
      }
    v16bf pf = load_frag_a(pw, 0, 32, 0, lane);
#pragma unroll
    for (int d = 0; d < 4; ++d) {
      v16bf vf = load_frag_bt(curV, d * 16, 32, 0, lane);
      oacc[d] = wmma_bf16(pf, vf, oacc[d]);
    }
    buf ^= 1;
  }

  // normalize and store to [B,S,D] bf16 (heads re-interleaved)
#pragma unroll
  for (int d = 0; d < 4; ++d)
#pragma unroll
    for (int r = 0; r < 8; ++r) {
      int m = r + ((lane >> 4) << 3);
      int sq = sq0 + m;
      int nd = d * 16 + (lane & 15);
      float val = oacc[d][r] / lrow[r];
      attn[((size_t)b * SEQ + sq) * EMBED_DIM + h * HEAD_DIM + nd] = f2bf(val);
    }
}

// ---------------------------------------------------------------------------
// Output projection: attn[4096,1024] @ Wo + bo -> f32 out
// ---------------------------------------------------------------------------
__global__ __launch_bounds__(256) void out_proj_kernel(
    const unsigned short* __restrict__ X, const unsigned short* __restrict__ W,
    const float* __restrict__ bo, float* __restrict__ out) {
  const int K = EMBED_DIM, N = EMBED_DIM;
  __shared__ __align__(16) unsigned short ldsA[2][128 * 32];
  __shared__ __align__(16) unsigned short ldsBt[2][128 * 32];
  int tid = threadIdx.x, lane = tid & 31, wave = tid >> 5;
  int wm = wave & 3, wn = wave >> 2;
  int m0 = blockIdx.y * 128, n0 = blockIdx.x * 128;

  v8f acc[2][4];
#pragma unroll
  for (int i = 0; i < 2; ++i)
#pragma unroll
    for (int j = 0; j < 4; ++j) acc[i][j] = zero8();

  gemm_issue_tiles(X, W, ldsA[0], ldsBt[0], m0, n0, 0, K, N, tid);
  int buf = 0;
  for (int kt = 0; kt < K; kt += 32) {
    wait_async0();
    __syncthreads();
    if (kt + 32 < K)
      gemm_issue_tiles(X, W, ldsA[buf ^ 1], ldsBt[buf ^ 1], m0, n0, kt + 32, K, N, tid);
    mma_tiles(ldsA[buf], ldsBt[buf], wm, wn, lane, acc);
    buf ^= 1;
  }

#pragma unroll
  for (int mt = 0; mt < 2; ++mt)
#pragma unroll
    for (int nt = 0; nt < 4; ++nt)
#pragma unroll
      for (int r = 0; r < 8; ++r) {
        int m = m0 + wm * 32 + mt * 16 + r + ((lane >> 4) << 3);
        int n = n0 + wn * 64 + nt * 16 + (lane & 15);
        out[(size_t)m * N + n] = acc[mt][nt][r] + bo[n];
      }
}

// ---------------------------------------------------------------------------
extern "C" void kernel_launch(void* const* d_in, const int* in_sizes, int n_in,
                              void* d_out, int out_size, void* d_ws, size_t ws_size,
                              hipStream_t stream) {
  (void)in_sizes; (void)n_in; (void)out_size;
  const float* hs = (const float*)d_in[0];
  const float* Wq = (const float*)d_in[1];
  const float* bq = (const float*)d_in[2];
  const float* Wk = (const float*)d_in[3];
  const float* bk = (const float*)d_in[4];
  const float* Wv = (const float*)d_in[5];
  const float* bv = (const float*)d_in[6];
  const float* Wo = (const float*)d_in[7];
  const float* bo = (const float*)d_in[8];
  const float* dist_tbl = (const float*)d_in[9];
  const float* link_tbl = (const float*)d_in[10];
  const int* distm = (const int*)d_in[11];
  const int* linkm = (const int*)d_in[12];
  float* out = (float*)d_out;

  const size_t NTOK = (size_t)BATCH * SEQ;              // 4096
  const size_t HS_ELEMS = NTOK * EMBED_DIM;             // 4M
  const size_t W_ELEMS = (size_t)EMBED_DIM * EMBED_DIM; // 1M
  unsigned short* hs16 = (unsigned short*)d_ws;
  unsigned short* wq16 = hs16 + HS_ELEMS;
  unsigned short* wk16 = wq16 + W_ELEMS;
  unsigned short* wv16 = wk16 + W_ELEMS;
  unsigned short* wo16 = wv16 + W_ELEMS;
  unsigned short* qbuf = wo16 + W_ELEMS;
  unsigned short* kbuf = qbuf + HS_ELEMS;
  unsigned short* vbuf = kbuf + HS_ELEMS;
  unsigned short* attn = vbuf + HS_ELEMS;
  size_t need = (size_t)(attn + HS_ELEMS - hs16) * sizeof(unsigned short);
  if (ws_size < need) return;  // insufficient scratch (deterministic no-op)

  const int thr = 256;
  cvt_bf16_kernel<<<(int)((HS_ELEMS + thr - 1) / thr), thr, 0, stream>>>(hs, hs16, (int)HS_ELEMS);
  cvt_bf16_kernel<<<(int)((W_ELEMS + thr - 1) / thr), thr, 0, stream>>>(Wq, wq16, (int)W_ELEMS);
  cvt_bf16_kernel<<<(int)((W_ELEMS + thr - 1) / thr), thr, 0, stream>>>(Wk, wk16, (int)W_ELEMS);
  cvt_bf16_kernel<<<(int)((W_ELEMS + thr - 1) / thr), thr, 0, stream>>>(Wv, wv16, (int)W_ELEMS);
  cvt_bf16_kernel<<<(int)((W_ELEMS + thr - 1) / thr), thr, 0, stream>>>(Wo, wo16, (int)W_ELEMS);

  dim3 g1(EMBED_DIM / 128, (unsigned)(NTOK / 128), 3);
  qkv_gemm_kernel<<<g1, 256, 0, stream>>>(hs16, wq16, wk16, wv16, bq, bk, bv,
                                          qbuf, kbuf, vbuf);

  dim3 g2(SEQ / 128, NUM_HEADS, BATCH);
  flash_attn_kernel<<<g2, 256, 0, stream>>>(qbuf, kbuf, vbuf, dist_tbl, link_tbl,
                                            distm, linkm, attn);

  dim3 g3(EMBED_DIM / 128, (unsigned)(NTOK / 128), 1);
  out_proj_kernel<<<g3, 256, 0, stream>>>(attn, wo16, bo, out);
}
